// vMF_69733089018370
// MI455X (gfx1250) — matrix operand order repaired and checked by reading, not compiled, not run on hardware
//
#include <hip/hip_runtime.h>
#include <hip/hip_bf16.h>

// ---------------- problem sizes ----------------
#define HID_K 1024
#define LAT_N 512
#define BATCH 16384

// ---------------- WMMA vector types ----------------
typedef __attribute__((ext_vector_type(16))) __bf16 v16bf;
typedef __attribute__((ext_vector_type(8)))  __bf16 v8bf;
typedef __attribute__((ext_vector_type(8)))  float  v8f;
typedef __attribute__((ext_vector_type(4)))  unsigned u32x4;
typedef __attribute__((ext_vector_type(8)))  unsigned u32x8;

// ---------------- tiling ----------------
// fused kernel: workgroup owns complete rows: 64(M) x 512(N), K-step 32.
// 8 waves = 4(M) x 2(N): wave = 16 rows x 256 cols = 16 WMMA C tiles.
constexpr int TM = 64;
constexpr int TK = 32;

// ---------------- gfx1250 async / tensor helpers ----------------
#if __has_builtin(__builtin_amdgcn_s_wait_asynccnt)
#define WAIT_ASYNC(n) __builtin_amdgcn_s_wait_asynccnt(n)
#else
#define WAIT_ASYNC(n) asm volatile("s_wait_asynccnt %0" :: "i"(n) : "memory")
#endif
#if __has_builtin(__builtin_amdgcn_s_wait_tensorcnt)
#define WAIT_TENSOR(n) __builtin_amdgcn_s_wait_tensorcnt(n)
#else
#define WAIT_TENSOR(n) asm volatile("s_wait_tensorcnt %0" :: "i"(n) : "memory")
#endif

__device__ __forceinline__ void async_b128_to_lds(const float* gsrc, const float* ldst) {
    // LDS offset = low 32 bits of the generic pointer (aperture bits discarded by HW)
    unsigned lds = (unsigned)(unsigned long long)(const void*)ldst;
    asm volatile("global_load_async_to_lds_b128 %0, %1, off"
                 :: "v"(lds), "v"(gsrc) : "memory");
}

// TDM: DMA a 2D tile (TK x TM f32 rows, row stride HID_K) from global -> LDS.
// D# group0: count=1 | lds_addr | global_addr[56:0] | type=2("image")
// D# group1: data_size=4B | tensor_dim0=HID_K | tensor_dim1=BATCH |
//            tile_dim0=TK | tile_dim1=TM | tensor_dim0_stride=HID_K
__device__ __forceinline__ void tdm_load_tile_f32(const float* gsrc, const float* ldst) {
    const unsigned long long ga = (unsigned long long)gsrc;
    const unsigned ldsOff = (unsigned)(unsigned long long)(const void*)ldst;
    u32x4 g0;
    g0[0] = 1u;                                            // count=1 (valid user D#)
    g0[1] = ldsOff;                                        // lds_addr
    g0[2] = (unsigned)(ga & 0xffffffffu);                  // global_addr[31:0]
    g0[3] = (unsigned)((ga >> 32) & 0x01ffffffu)           // global_addr[56:32]
          | (2u << 30);                                    // type=2
    u32x8 g1;
    g1[0] = (2u << 16);                                    // data_size=2 (4 bytes)
    g1[1] = ((unsigned)HID_K & 0xffffu) << 16;             // tensor_dim0[15:0]
    g1[2] = ((unsigned)HID_K >> 16)                        // tensor_dim0[31:16]
          | (((unsigned)BATCH & 0xffffu) << 16);           // tensor_dim1[15:0]
    g1[3] = ((unsigned)BATCH >> 16)                        // tensor_dim1[31:16]
          | ((unsigned)TK << 16);                          // tile_dim0=32
    g1[4] = (unsigned)TM;                                  // tile_dim1=64, tile_dim2=0
    g1[5] = (unsigned)HID_K;                               // tensor_dim0_stride[31:0]
    g1[6] = 0u;                                            // stride hi | dim1_stride lo
    g1[7] = 0u;
    asm volatile("tensor_load_to_lds %0, %1" :: "s"(g0), "s"(g1) : "memory");
}

// ---------------- prep: W[k][n] f32 -> Wb[n][k] bf16 (WMMA-B friendly) ----------------
__global__ __launch_bounds__(256)
void vmf_prep_w_kernel(const float* __restrict__ W, __bf16* __restrict__ Wb)
{
    const int idx = blockIdx.x * 256 + threadIdx.x;   // 131072 threads
    const int n   = idx >> 8;                         // 0..511
    const int k4  = (idx & 255) * 4;                  // 0..1020 step 4
    __bf16* dst = &Wb[(size_t)n * HID_K + k4];
    #pragma unroll
    for (int j = 0; j < 4; ++j)
        dst[j] = (__bf16)W[(size_t)(k4 + j) * LAT_N + n];
}

// ---------------- RNG helpers (counter-based, deterministic) ----------------
__device__ __forceinline__ unsigned long long mix64(unsigned long long x) {
    x += 0x9E3779B97F4A7C15ull;
    x = (x ^ (x >> 30)) * 0xBF58476D1CE4E5B9ull;
    x = (x ^ (x >> 27)) * 0x94D049BB133111EBull;
    return x ^ (x >> 31);
}
__device__ __forceinline__ float u01(unsigned long long ctr, unsigned long long key) {
    unsigned long long h = mix64(ctr ^ mix64(key));
    return ((float)(unsigned)(h >> 40) + 0.5f) * (1.0f / 16777216.0f);  // (0,1)
}
// Marsaglia-Tsang Gamma(a), a >> 1 (accept prob ~0.998 at a=255.5)
__device__ __forceinline__ float sample_gamma(float a, unsigned long long& ctr,
                                              unsigned long long key) {
    const float d = a - (1.0f / 3.0f);
    const float c = rsqrtf(9.0f * d);
    for (int it = 0; it < 8; ++it) {
        float u1 = fmaxf(u01(ctr++, key), 1.0e-12f);
        float u2 = u01(ctr++, key);
        float u3 = fmaxf(u01(ctr++, key), 1.0e-12f);
        float x  = sqrtf(-2.0f * logf(u1)) * __cosf(6.28318530718f * u2);
        float v  = 1.0f + c * x;
        if (v <= 0.0f) continue;
        v = v * v * v;
        if (logf(u3) < 0.5f * x * x + d - d * v + d * logf(v)) return d * v;
    }
    return a;
}
__device__ __forceinline__ float wave_sum(float v) {
    #pragma unroll
    for (int off = 16; off > 0; off >>= 1) v += __shfl_xor(v, off, 32);
    return v;
}
__device__ __forceinline__ v16bf cvt16(float4 a, float4 b, float4 c, float4 d) {
    v16bf r;
    r[0]  = (__bf16)a.x; r[1]  = (__bf16)a.y; r[2]  = (__bf16)a.z; r[3]  = (__bf16)a.w;
    r[4]  = (__bf16)b.x; r[5]  = (__bf16)b.y; r[6]  = (__bf16)b.z; r[7]  = (__bf16)b.w;
    r[8]  = (__bf16)c.x; r[9]  = (__bf16)c.y; r[10] = (__bf16)c.z; r[11] = (__bf16)c.w;
    r[12] = (__bf16)d.x; r[13] = (__bf16)d.y; r[14] = (__bf16)d.z; r[15] = (__bf16)d.w;
    return r;
}

// ---------------- fused GEMM + normalize + vMF sampler ----------------
__global__ __launch_bounds__(256)
void vmf_fused_kernel(const float* __restrict__ A,
                      const __bf16* __restrict__ Wb,   // [512][1024] bf16
                      const float* __restrict__ bias,
                      float* __restrict__ out)
{
    __shared__ __align__(16) float sAf[2][TM][TK];     // 16 KB, TDM-filled f32
    __shared__ __align__(16) float sMu[TM][LAT_N];     // 128 KB row buffer
    __shared__ __align__(16) float sBias[LAT_N];       // 2 KB, async-filled

    const int tid    = threadIdx.x;
    const int lane   = tid & 31;
    const int wid    = tid >> 5;
    const int waveM  = wid & 3;        // 0..3 -> 16-row slab
    const int waveN  = wid >> 2;       // 0..1 -> 256-col half
    const int row16  = lane & 15;
    const int laneHi = lane >> 4;
    const int mBase  = blockIdx.x * TM;

    // one-time: stage bias through the async global->LDS path (ASYNCcnt)
    if (tid < LAT_N / 4)
        async_b128_to_lds(&bias[tid * 4], &sBias[tid * 4]);

    v8f acc[16];
    #pragma unroll
    for (int t = 0; t < 16; ++t)
        acc[t] = (v8f){0.f,0.f,0.f,0.f,0.f,0.f,0.f,0.f};

    // B fragment loader: straight from L2-resident bf16 Wb[n][k]
    const __bf16* bWave = Wb + (size_t)(waveN * 256 + row16) * HID_K + laneHi * 8;
    auto load_bF = [&](int t, int i) -> v16bf {
        const __bf16* bp = bWave + (size_t)t * 16 * HID_K + i * TK;
        v8bf b0 = *(const v8bf*)bp;
        v8bf b1 = *(const v8bf*)(bp + 16);
        return __builtin_shufflevector(b0, b1,
                 0,1,2,3,4,5,6,7,8,9,10,11,12,13,14,15);
    };

    // TDM double-buffered prologue: wave 0 drives the tensor DMA engine
    if (wid == 0) {
        tdm_load_tile_f32(&A[(size_t)mBase * HID_K + 0 * TK], &sAf[0][0][0]);
        tdm_load_tile_f32(&A[(size_t)mBase * HID_K + 1 * TK], &sAf[1][0][0]);
    }

    for (int i = 0; i < HID_K / TK; ++i) {              // 32 K-steps
        if (wid == 0) {                                  // tile i resident (in-order)
            if (i < HID_K / TK - 1) { WAIT_TENSOR(1); }
            else                    { WAIT_TENSOR(0); }
        }
        __syncthreads();                                 // visible to all waves

        // ---- A fragment: 16-bit A 16x32 layout from f32 LDS (convert here) ----
        const float* aRowP = &sAf[i & 1][waveM * 16 + row16][0];
        const int kb = laneHi * 8;
        const float4 a0 = *(const float4*)&aRowP[kb];
        const float4 a1 = *(const float4*)&aRowP[kb + 4];
        const float4 a2 = *(const float4*)&aRowP[kb + 16];
        const float4 a3 = *(const float4*)&aRowP[kb + 20];
        const v16bf aF = cvt16(a0, a1, a2, a3);

        // ---- 16 WMMAs, B fragments double-buffered so a load stays in flight ----
        v16bf bF = load_bF(0, i);
        #pragma unroll
        for (int t = 0; t < 16; ++t) {
            v16bf bN = bF;
            if (t < 15) bN = load_bF(t + 1, i);          // issue next load first
            acc[t] = __builtin_amdgcn_wmma_f32_16x16x32_bf16(
                         false, aF, false, bF, (short)0, acc[t], false, false);
            bF = bN;
        }

        __syncthreads();                                 // done reading buf (i&1)
        if (wid == 0 && i + 2 < HID_K / TK)
            tdm_load_tile_f32(&A[(size_t)mBase * HID_K + (i + 2) * TK],
                              &sAf[i & 1][0][0]);
    }

    // bias staged long ago; make it visible before the epilogue
    WAIT_ASYNC(0);
    __syncthreads();

    // ---- epilogue: +bias -> LDS row buffer (C layout: VGPR r -> M=r / M=8+r) ----
    const int mLoc0 = waveM * 16 + laneHi * 8;
    #pragma unroll
    for (int t = 0; t < 16; ++t) {
        const int n = waveN * 256 + t * 16 + row16;
        const float bb = sBias[n];
        #pragma unroll
        for (int r = 0; r < 8; ++r)
            sMu[mLoc0 + r][n] = acc[t][r] + bb;
    }
    __syncthreads();

    // ---- phase 2: wave-per-row vMF sampling, 8 rows per wave ----
    #pragma unroll 1
    for (int rr = 0; rr < 8; ++rr) {
        const int lrow = wid * 8 + rr;
        const int row  = mBase + lrow;

        float mu[16];
        float ss = 0.f;
        #pragma unroll
        for (int j = 0; j < 16; ++j) {
            mu[j] = sMu[lrow][j * 32 + lane];
            ss += mu[j] * mu[j];
        }
        ss = wave_sum(ss);
        const float rinv = rsqrtf(fmaxf(ss, 1.0e-30f));
        #pragma unroll
        for (int j = 0; j < 16; ++j) mu[j] *= rinv;

        // Wood's rejection sampling for w (lane 0, broadcast)
        float w = 0.f;
        if (lane == 0) {
            const float KAP  = 1.0f;
            const float DIMf = (float)(LAT_N - 1);
            const float b0 = DIMf / (sqrtf(4.0f * KAP * KAP + DIMf * DIMf) + 2.0f * KAP);
            const float x0 = (1.0f - b0) / (1.0f + b0);
            const float c0 = KAP * x0 + DIMf * logf(1.0f - x0 * x0);
            const float a  = 0.5f * DIMf;
            unsigned long long key = 42ull;
            unsigned long long ctr = ((unsigned long long)row) << 20;
            bool done = false;
            for (int rj = 0; rj < 64 && !done; ++rj) {
                float g1 = sample_gamma(a, ctr, key);
                float g2 = sample_gamma(a, ctr, key);
                float z  = g1 / (g1 + g2);                       // Beta(a,a)
                float wp = (1.0f - (1.0f + b0) * z) / (1.0f - (1.0f - b0) * z);
                float u  = u01(ctr++, key);
                if (KAP * wp + DIMf * log1pf(-x0 * wp) - c0 >= logf(u)) {
                    w = wp; done = true;
                }
            }
        }
        w = __shfl(w, 0, 32);

        // tangent Gaussian, orthogonalize, renormalize, combine
        float v[16];
        float dot = 0.f;
        const unsigned long long key2 = 0x243F6A8885A308D3ull ^ 42ull;
        #pragma unroll
        for (int j = 0; j < 16; ++j) {
            unsigned long long c = ((unsigned long long)row << 10) + (unsigned)(j * 32 + lane);
            float u1 = fmaxf(u01(2 * c, key2), 1.0e-12f);
            float u2 = u01(2 * c + 1, key2);
            v[j] = sqrtf(-2.0f * logf(u1)) * __cosf(6.28318530718f * u2);
            dot += mu[j] * v[j];
        }
        dot = wave_sum(dot);
        float osum = 0.f;
        #pragma unroll
        for (int j = 0; j < 16; ++j) { v[j] -= mu[j] * dot; osum += v[j] * v[j]; }
        osum = wave_sum(osum);
        const float oinv = rsqrtf(fmaxf(osum, 1.0e-30f));
        const float sw   = sqrtf(fmaxf(0.0f, 1.0f - w * w));

        float* orow = &out[(size_t)row * LAT_N];
        #pragma unroll
        for (int j = 0; j < 16; ++j)
            orow[j * 32 + lane] = v[j] * oinv * sw + mu[j] * w;
    }
}

extern "C" void kernel_launch(void* const* d_in, const int* in_sizes, int n_in,
                              void* d_out, int out_size, void* d_ws, size_t ws_size,
                              hipStream_t stream) {
    (void)in_sizes; (void)n_in; (void)out_size; (void)ws_size;
    const float* A    = (const float*)d_in[0];   // [16384,1024]
    const float* W    = (const float*)d_in[1];   // [1024,512]
    const float* bias = (const float*)d_in[2];   // [512]
    // d_in[3] = n_sample (always 1 for this harness)
    __bf16* Wb = (__bf16*)d_ws;                  // [512,1024] bf16 = 1 MB scratch
    float* out = (float*)d_out;

    vmf_prep_w_kernel<<<512, 256, 0, stream>>>(W, Wb);
    vmf_fused_kernel<<<BATCH / TM, 256, 0, stream>>>(A, Wb, bias, out);
}
